// InstanceSegAlgoFPN_JIT_25074019074129
// MI455X (gfx1250) — compile-verified, matrix operand order checked by reading, not run on hardware
//
#include <hip/hip_runtime.h>
#include <stdint.h>

// ---------------- problem constants (match reference) ----------------
#define B_        4
#define N_        2048
#define C_        10
#define NC_       (N_ * C_)      // 20480 candidates per batch
#define KNMS_     1024
#define SORTN_    32768          // next pow2 >= NC_
#define MAXP_     100
#define ROI_      14
#define CF_       256
#define IMGSZ_    1024.0f
#define NMS_THR_  0.3f
#define SCORE_THR_ 0.1f
#define SCALE_CLIP_ 4.135166556742356f   // ln(1000/16)

#define NPROP_    (B_ * N_)      // 8192 proposals
#define NWAVE_    (NPROP_ / 16)  // 512 WMMA tiles (16 rows each)

typedef __attribute__((ext_vector_type(16))) _Float16 v16h;
typedef __attribute__((ext_vector_type(8)))  float    v8f;

// =====================================================================
// Stage 1a: per-proposal exp() with hi/lo f16 split, written into the
// WMMA A-tile wave layout (ISA 7.12.2, 16-bit A 16x32). hi at columns
// 0..11, lo residual at columns 16..27, B = ones => f32 accumulator
// reproduces the f32 row sum to ~1e-7 relative.
// =====================================================================
__global__ __launch_bounds__(256)
void stage_kernel(const float* __restrict__ cls,
                  _Float16* __restrict__ wsA,
                  _Float16* __restrict__ wsB) {
  const int p = blockIdx.x * 256 + threadIdx.x;   // proposal 0..8191
  const int w = p >> 4;                           // wave tile
  const int m = p & 15;                           // row in tile

  const float* lg = cls + (size_t)p * (C_ + 1);
  float rmax = lg[0];
#pragma unroll
  for (int j = 1; j < C_ + 1; ++j) rmax = fmaxf(rmax, lg[j]);

  _Float16* A0 = wsA + ((size_t)w * 32 + m) * 16;        // lane m
  _Float16* A1 = wsA + ((size_t)w * 32 + 16 + m) * 16;   // lane m+16
#pragma unroll
  for (int i = 0; i < 8; ++i) {
    const float v0 = __expf(lg[i] - rmax);               // cols 0..7
    const _Float16 h0 = (_Float16)v0;
    A0[i]     = h0;
    A0[8 + i] = (_Float16)(v0 - (float)h0);
    const int c1 = 8 + i;                                // cols 8..15
    const float v1 = (c1 <= C_) ? __expf(lg[c1] - rmax) : 0.0f;
    const _Float16 h1 = (_Float16)v1;
    A1[i]     = h1;
    A1[8 + i] = (_Float16)(v1 - (float)h1);
  }
  if (p < 512) wsB[p] = (_Float16)1.0f;                  // ones B-tile
}

// =====================================================================
// Stage 1b: softmax row-sums on the matrix core (probe-verified WMMA
// pattern: whole-vector loads -> one builtin call -> vector store).
// =====================================================================
__global__ void wmma_rowsum_kernel(const _Float16* __restrict__ A,
                                   const _Float16* __restrict__ Bm,
                                   float* __restrict__ D) {
  const int g = blockIdx.x * 32 + threadIdx.x;   // global lane
  v16h a = *(const v16h*)(A + (size_t)g * 16);
  v16h b = *(const v16h*)(Bm + (size_t)threadIdx.x * 16);
  v8f  c = {};
  c = __builtin_amdgcn_wmma_f32_16x16x32_f16(
      false, a, false, b, (short)0, c, false, false);
  *(v8f*)(D + (size_t)g * 8) = c;
}

// =====================================================================
// Stage 1c-box: box decode only (FMA-heavy float math + b32 stores;
// no comparisons/selects, no 64-bit integer ops).
// =====================================================================
__global__ __launch_bounds__(256)
void decode_box_kernel(const float* __restrict__ proposals,
                       const float* __restrict__ bbx,
                       float* __restrict__ boxes_all) {
  const int p = blockIdx.x * 256 + threadIdx.x;   // proposal 0..8191

  const float* pr = proposals + (size_t)p * 4;
  const float py0 = pr[0], px0 = pr[1], py1 = pr[2], px1 = pr[3];
  const float cy = 0.5f * (py0 + py1), cx = 0.5f * (px0 + px1);
  const float hy = py1 - py0,          hx = px1 - px0;

  float* bb = boxes_all + (size_t)p * C_ * 4;     // [p][c][4] layout

#pragma unroll
  for (int c = 0; c < C_; ++c) {
    const float* dl = bbx + ((size_t)p * C_ + c) * 4;
    const float dy = dl[0] * 0.1f, dx = dl[1] * 0.1f;   // REG_W = 10
    const float dh = dl[2] * 0.2f, dw = dl[3] * 0.2f;   // REG_W = 5
    const float oy = cy + hy * dy, ox = cx + hx * dx;
    const float hh = 0.5f * hy * __expf(fminf(dh, SCALE_CLIP_));
    const float hw = 0.5f * hx * __expf(fminf(dw, SCALE_CLIP_));
    bb[c * 4 + 0] = fminf(fmaxf(oy - hh, 0.0f), IMGSZ_);
    bb[c * 4 + 1] = fminf(fmaxf(ox - hw, 0.0f), IMGSZ_);
    bb[c * 4 + 2] = fminf(fmaxf(oy + hh, 0.0f), IMGSZ_);
    bb[c * 4 + 3] = fminf(fmaxf(ox + hw, 0.0f), IMGSZ_);
  }
}

// =====================================================================
// Stage 1c-score: scores + validity + 32-bit sort key (high word only;
// the index low word is attached inside sort_kernel, keeping ALL
// 64-bit integer ops out of this kernel).
// khi = valid ? ~(bits(sc) | 0x80000000) : 0xFFFFFFFF
// ascending (khi, index) == score desc, index asc, invalid last.
// =====================================================================
__global__ __launch_bounds__(256)
void decode_score_kernel(const float* __restrict__ cls,
                         const float* __restrict__ Dsum,
                         const float* __restrict__ boxes_all,
                         unsigned* __restrict__ keyhi) {
  const int p = blockIdx.x * 256 + threadIdx.x;   // proposal 0..8191
  const int w = p >> 4;
  const int m = p & 15;

  // D layout (ISA 7.12.2 C/D 16x16): VGPR r of lane 0 = row r,
  // VGPR r of lane 16 = row r+8.
  const float denom = (m < 8)
      ? Dsum[((size_t)w * 32 + 0)  * 8 + m]
      : Dsum[((size_t)w * 32 + 16) * 8 + (m - 8)];
  const float rden = __builtin_amdgcn_rcpf(denom);   // v_rcp_f32, ~1 ulp

  const float* lg = cls + (size_t)p * (C_ + 1);
  float rmax = lg[0];
#pragma unroll
  for (int j = 1; j < C_ + 1; ++j) rmax = fmaxf(rmax, lg[j]);

  const float* bb = boxes_all + (size_t)p * C_ * 4;
  unsigned* kh = keyhi + (size_t)p * C_;

#pragma unroll
  for (int c = 0; c < C_; ++c) {
    const float sc = __expf(lg[c + 1] - rmax) * rden;
    const float b0 = bb[c * 4 + 0], b1 = bb[c * 4 + 1];
    const float b2 = bb[c * 4 + 2], b3 = bb[c * 4 + 3];
    const bool valid = (sc > SCORE_THR_) && (b2 > b0) && (b3 > b1);
    // sc > 0 when valid => sign bit 0 => bits are order-preserving.
    const unsigned khi = valid ? ~(__float_as_uint(sc) | 0x80000000u)
                               : 0xFFFFFFFFu;
    kh[c] = khi;
  }
}

// =====================================================================
// Kernel 2: exact top-1024 per batch via full bitonic sort of 32768
// (khi, index) u64 keys in 256 KB of dynamic LDS (CDNA5: 320 KB/WGP).
// One block per batch.
// =====================================================================
__global__ __launch_bounds__(1024)
void sort_kernel(const unsigned* __restrict__ keyhi,
                 const float* __restrict__ boxes_all,
                 float* __restrict__ cand_boxes,
                 float* __restrict__ cand_scores,
                 int* __restrict__ cand_cls) {
  extern __shared__ unsigned long long s[];
  const int tid = threadIdx.x;
  const int b   = blockIdx.x;
  const unsigned* kh = keyhi + (size_t)b * NC_;
  for (int i = tid; i < SORTN_; i += 1024)
    s[i] = (i < NC_)
        ? (((unsigned long long)kh[i] << 32) | (unsigned)i)
        : ~0ull;                                 // pad sorts last
  __syncthreads();
  for (unsigned k = 2; k <= SORTN_; k <<= 1) {
    for (unsigned j = k >> 1; j > 0; j >>= 1) {
      for (unsigned i = tid; i < SORTN_; i += 1024) {
        const unsigned ixj = i ^ j;
        if (ixj > i) {
          const unsigned long long a = s[i], c = s[ixj];
          const bool up = ((i & k) == 0);
          if ((a > c) == up) { s[i] = c; s[ixj] = a; }
        }
      }
      __syncthreads();
    }
  }
  if (tid < KNMS_) {
    const unsigned long long key = s[tid];
    const unsigned khi = (unsigned)(key >> 32);
    const float sc = (khi == 0xFFFFFFFFu)
        ? -1.0f
        : __uint_as_float(~khi & 0x7FFFFFFFu);
    const int j = (int)(unsigned)key;            // < NC_ for all top-1024
    cand_scores[b * KNMS_ + tid] = sc;
    cand_cls[b * KNMS_ + tid]    = j % C_;
    const float* src = boxes_all + ((size_t)b * NC_ + j) * 4;
    float* dst = cand_boxes + ((size_t)b * KNMS_ + tid) * 4;
    dst[0] = src[0]; dst[1] = src[1]; dst[2] = src[2]; dst[3] = src[3];
  }
}

// =====================================================================
// Kernel 3: greedy NMS over sorted candidates (LDS-resident) + exact
// top-100 select + FPN level assignment. One 1024-thread block/batch.
// =====================================================================
__global__ __launch_bounds__(1024)
void nms_finalize_kernel(const float* __restrict__ cand_boxes,
                         const float* __restrict__ cand_scores,
                         const int* __restrict__ cand_cls,
                         float* __restrict__ out,
                         int* __restrict__ lvl_ws,
                         int* __restrict__ valid_ws) {
  __shared__ float sb[KNMS_ * 4];
  __shared__ float ss[KNMS_];
  __shared__ int   skeep[KNMS_];
  __shared__ int   ssel[MAXP_];
  const int tid = threadIdx.x;
  const int b   = blockIdx.x;

  const float* cb = cand_boxes + (size_t)b * KNMS_ * 4;
  for (int i = tid; i < KNMS_ * 4; i += 1024) sb[i] = cb[i];
  ss[tid]    = cand_scores[b * KNMS_ + tid];
  skeep[tid] = 1;
  __syncthreads();

  const float y0 = sb[tid * 4 + 0], x0 = sb[tid * 4 + 1];
  const float y1 = sb[tid * 4 + 2], x1 = sb[tid * 4 + 3];
  const float area_t = (y1 - y0) * (x1 - x0);

  for (int i = 0; i < KNMS_ - 1; ++i) {
    __syncthreads();
    if (!skeep[i]) continue;                 // uniform (LDS broadcast)
    if (tid > i && skeep[tid]) {
      const float iy0 = sb[i * 4 + 0], ix0 = sb[i * 4 + 1];
      const float iy1 = sb[i * 4 + 2], ix1 = sb[i * 4 + 3];
      const float ai  = (iy1 - iy0) * (ix1 - ix0);
      const float wy  = fmaxf(fminf(iy1, y1) - fmaxf(iy0, y0), 0.0f);
      const float wx  = fmaxf(fminf(ix1, x1) - fmaxf(ix0, x0), 0.0f);
      const float inter = wy * wx;
      const float iou =
          inter * __builtin_amdgcn_rcpf(ai + area_t - inter + 1e-6f);
      if (iou > NMS_THR_) skeep[tid] = 0;
    }
  }
  __syncthreads();

  if (tid == 0) {
    int cnt = 0;
    for (int i = 0; i < KNMS_ && cnt < MAXP_; ++i)
      if (skeep[i] && ss[i] > 0.0f) ssel[cnt++] = i;
    for (int i = 0; i < KNMS_ && cnt < MAXP_; ++i)
      if (!(skeep[i] && ss[i] > 0.0f)) ssel[cnt++] = i;
  }
  __syncthreads();

  if (tid < MAXP_) {
    const int i = ssel[tid];
    const float fy0 = sb[i * 4 + 0], fx0 = sb[i * 4 + 1];
    const float fy1 = sb[i * 4 + 2], fx1 = sb[i * 4 + 3];
    const float fsc = (skeep[i] && ss[i] > 0.0f) ? ss[i] : -1.0f;
    const int fvalid = (fsc > 0.0f) ? 1 : 0;
    const int slot = b * MAXP_ + tid;
    float* obox = out;                         // [0, 1600)
    float* osc  = out + B_ * MAXP_ * 4;        // [1600, 2000)
    float* ocls = osc + B_ * MAXP_;            // [2000, 2400)
    float* oval = ocls + B_ * MAXP_;           // [2400, 2800)
    obox[slot * 4 + 0] = fy0; obox[slot * 4 + 1] = fx0;
    obox[slot * 4 + 2] = fy1; obox[slot * 4 + 3] = fx1;
    osc[slot]  = fsc;
    ocls[slot] = (float)cand_cls[b * KNMS_ + i];
    oval[slot] = fvalid ? 1.0f : 0.0f;
    const float hh = fy1 - fy0, ww = fx1 - fx0;
    const float scale = __builtin_amdgcn_sqrtf(fmaxf(hh * ww, 1e-6f));
    // v_log_f32 is log2 -- matches reference jnp.log2
    const float tl =
        floorf(2.0f + __builtin_amdgcn_logf(scale * (1.0f / 224.0f) + 1e-6f));
    lvl_ws[slot]   = (int)fminf(fmaxf(tl, 0.0f), 3.0f);
    valid_ws[slot] = fvalid;
  }
}

// =====================================================================
// Kernel 4: ROI-Align at the assigned FPN level only. One block per
// (batch, roi); sample weights/indices precomputed once in LDS;
// channel-major coalesced writes; 4 gathers per output.
// =====================================================================
__global__ __launch_bounds__(256)
void roi_kernel(const float* __restrict__ f0, const float* __restrict__ f1,
                const float* __restrict__ f2, const float* __restrict__ f3,
                const float* __restrict__ out_base,
                const int* __restrict__ lvl_ws,
                const int* __restrict__ valid_ws,
                float* __restrict__ rois) {
  const int slot = blockIdx.x;                 // 0..399
  const int b    = slot / MAXP_;
  const int tid  = threadIdx.x;
  float* dst = rois + (size_t)slot * CF_ * ROI_ * ROI_;

  if (!valid_ws[slot]) {
    for (int i = tid; i < CF_ * ROI_ * ROI_; i += 256) dst[i] = 0.0f;
    return;
  }

  const int lvl = lvl_ws[slot];
  const float* f; int H;
  if      (lvl == 0) { f = f0; H = 256; }
  else if (lvl == 1) { f = f1; H = 128; }
  else if (lvl == 2) { f = f2; H = 64;  }
  else               { f = f3; H = 32;  }
  const int W = H;
  const float sy = (float)H * (1.0f / IMGSZ_);   // exact pow2 scale
  const float* bx = out_base + (size_t)slot * 4;
  const float py0 = (bx[0] - 0.5f) * sy + 0.5f;
  const float px0 = (bx[1] - 0.5f) * sy + 0.5f;
  const float py1 = (bx[2] - 0.5f) * sy + 0.5f;
  const float px1 = (bx[3] - 0.5f) * sy + 0.5f;

  __shared__ int   s_yi[ROI_], s_yh[ROI_], s_xi[ROI_], s_xh[ROI_];
  __shared__ float s_wy[ROI_], s_wx[ROI_];
  if (tid < ROI_) {
    float ys = py0 + ((float)tid + 0.5f) * (py1 - py0) * (1.0f / ROI_);
    ys = fminf(fmaxf(ys, 0.0f), (float)(H - 1));
    const float yl = floorf(ys);
    const int yi = (int)yl;
    s_yi[tid] = yi;
    s_yh[tid] = (yi + 1 < H - 1) ? yi + 1 : H - 1;
    s_wy[tid] = ys - yl;
  } else if (tid < 2 * ROI_) {
    const int k = tid - ROI_;
    float xs = px0 + ((float)k + 0.5f) * (px1 - px0) * (1.0f / ROI_);
    xs = fminf(fmaxf(xs, 0.0f), (float)(W - 1));
    const float xl = floorf(xs);
    const int xi = (int)xl;
    s_xi[k] = xi;
    s_xh[k] = (xi + 1 < W - 1) ? xi + 1 : W - 1;
    s_wx[k] = xs - xl;
  }
  __syncthreads();

  const float* fb = f + (size_t)b * CF_ * H * W;
  for (int idx = tid; idx < CF_ * ROI_ * ROI_; idx += 256) {
    const int c = idx / (ROI_ * ROI_);
    const int s = idx - c * (ROI_ * ROI_);
    const int y = s / ROI_, x = s - y * ROI_;
    const float* fc = fb + (size_t)c * H * W;
    const int yi = s_yi[y], yh = s_yh[y], xi = s_xi[x], xh = s_xh[x];
    const float wy = s_wy[y], wx = s_wx[x];
    const float v00 = fc[yi * W + xi], v01 = fc[yi * W + xh];
    const float v10 = fc[yh * W + xi], v11 = fc[yh * W + xh];
    dst[idx] = v00 * (1.0f - wy) * (1.0f - wx)
             + v01 * (1.0f - wy) * wx
             + v10 * wy * (1.0f - wx)
             + v11 * wy * wx;
  }
}

// =====================================================================
extern "C" void kernel_launch(void* const* d_in, const int* in_sizes, int n_in,
                              void* d_out, int out_size, void* d_ws, size_t ws_size,
                              hipStream_t stream) {
  const float* proposals = (const float*)d_in[0];
  const float* bbx       = (const float*)d_in[1];
  const float* cls       = (const float*)d_in[2];
  const float* f0        = (const float*)d_in[3];
  const float* f1        = (const float*)d_in[4];
  const float* f2        = (const float*)d_in[5];
  const float* f3        = (const float*)d_in[6];
  float* out = (float*)d_out;

  char* ws = (char*)d_ws;
  size_t off = 0;
  auto take = [&](size_t bytes) {
    char* p = ws + off;
    off = (off + bytes + 255) & ~(size_t)255;
    return p;
  };
  unsigned* keyhi    = (unsigned*)take((size_t)B_ * NC_ * sizeof(unsigned));
  float* boxes_all   = (float*)take((size_t)B_ * NC_ * 4 * sizeof(float));
  float* cand_boxes  = (float*)take((size_t)B_ * KNMS_ * 4 * sizeof(float));
  float* cand_scores = (float*)take((size_t)B_ * KNMS_ * sizeof(float));
  int*   cand_cls    = (int*)take((size_t)B_ * KNMS_ * sizeof(int));
  int*   lvl_ws      = (int*)take((size_t)B_ * MAXP_ * sizeof(int));
  int*   valid_ws    = (int*)take((size_t)B_ * MAXP_ * sizeof(int));
  _Float16* wsA      = (_Float16*)take((size_t)NWAVE_ * 32 * 16 * sizeof(_Float16));
  _Float16* wsB      = (_Float16*)take((size_t)512 * sizeof(_Float16));
  float*    wsD      = (float*)take((size_t)NWAVE_ * 32 * 8 * sizeof(float));
  (void)in_sizes; (void)n_in; (void)out_size; (void)ws_size;

  stage_kernel<<<NPROP_ / 256, 256, 0, stream>>>(cls, wsA, wsB);
  wmma_rowsum_kernel<<<NWAVE_, 32, 0, stream>>>(wsA, wsB, wsD);
  decode_box_kernel<<<NPROP_ / 256, 256, 0, stream>>>(proposals, bbx,
                                                      boxes_all);
  decode_score_kernel<<<NPROP_ / 256, 256, 0, stream>>>(cls, wsD, boxes_all,
                                                        keyhi);
  sort_kernel<<<B_, 1024, SORTN_ * sizeof(unsigned long long), stream>>>(
      keyhi, boxes_all, cand_boxes, cand_scores, cand_cls);
  nms_finalize_kernel<<<B_, 1024, 0, stream>>>(cand_boxes, cand_scores,
                                               cand_cls, out, lvl_ws, valid_ws);
  roi_kernel<<<B_ * MAXP_, 256, 0, stream>>>(f0, f1, f2, f3, out, lvl_ws,
                                             valid_ws, out + 2800);
}
